// SEQ2SEQ_42537356099856
// MI455X (gfx1250) — matrix operand order repaired and checked by reading, not compiled, not run on hardware
//
#include <hip/hip_runtime.h>
#include <hip/hip_bf16.h>
#include <math.h>

// Problem constants (from the reference)
#define B_    128
#define H_    512
#define E_    512
#define F_    4096
#define V_    4096
#define NIN   80
#define NOUT  25
#define TTOT  105          // NIN + NOUT
#define BOS_  (V_ - 3)
#define LEAK_ 0.01f

typedef unsigned short u16;
typedef __bf16 bf16t;
typedef bf16t v16bf __attribute__((ext_vector_type(16)));
typedef float v8f   __attribute__((ext_vector_type(8)));

union Frag { uint4 u[2]; u16 s[16]; v16bf v; };

__device__ __forceinline__ u16 f2bf(float f) {
  unsigned int u = __float_as_uint(f);
  unsigned int r = u + 0x7FFFu + ((u >> 16) & 1u);   // round-to-nearest-even
  return (u16)(r >> 16);
}

// ---------------------------------------------------------------------------
// bf16 WMMA GEMM:  C[M,N] = act(A[M,K] @ W[N,K]^T + bias)
//   A, W: bf16 row-major.  One wave -> 16(M) x 64(N) tile: 4x
//   v_wmma_f32_16x16x32_bf16 per 32-wide K-chunk reusing one A fragment.
//   All 5 fragment loads (1xA + 4xB) are issued before the WMMA burst so the
//   scheduler can use partial s_wait_loadcnt values and keep loads in flight.
//   8 waves/block stack in M -> 128 x 64 block tile (M=batch=128 exactly for
//   the sequential-step GEMMs).  Requires M%16==0, N%64==0, K%32==0.
// ---------------------------------------------------------------------------
template<int ACT>
__global__ __launch_bounds__(256) void k_gemm_wmma(
    const u16* __restrict__ Ap, int lda,
    const u16* __restrict__ W, int ldw,
    const float* __restrict__ bias,
    float* __restrict__ Cf, u16* __restrict__ Cb, int ldc,
    int M, int N, int K)
{
  const int wave  = threadIdx.x >> 5;
  const int lane  = threadIdx.x & 31;
  const int mtile = blockIdx.y * 128 + wave * 16;
  const int ntile = blockIdx.x * 64;
  if (mtile >= M) return;                         // wave-uniform
  const int mrow = mtile + (lane & 15);
  const int kbA  = (lane >> 4) * 16;              // A frag: lanes 0-15 K {0..15}
  const int ncol = ntile + (lane & 15);           // B frag: lane -> N
  const int kbB  = (lane >> 4) * 16;              // B frag: lane half -> K half

  const u16* Arow = Ap + (size_t)mrow * lda + kbA;
  const u16* W0 = W + (size_t)(ncol     ) * ldw + kbB;
  const u16* W1 = W + (size_t)(ncol + 16) * ldw + kbB;
  const u16* W2 = W + (size_t)(ncol + 32) * ldw + kbB;
  const u16* W3 = W + (size_t)(ncol + 48) * ldw + kbB;

  v8f acc[4] = {};

  for (int k0 = 0; k0 < K; k0 += 32) {
    Frag a, b0, b1, b2, b3;
    // Issue all loads up front (in-order completion -> partial waits possible)
    a.u[0]  = *(const uint4*)(Arow + k0);
    a.u[1]  = *(const uint4*)(Arow + k0 + 8);
    b0.u[0] = *(const uint4*)(W0 + k0);
    b0.u[1] = *(const uint4*)(W0 + k0 + 8);
    b1.u[0] = *(const uint4*)(W1 + k0);
    b1.u[1] = *(const uint4*)(W1 + k0 + 8);
    b2.u[0] = *(const uint4*)(W2 + k0);
    b2.u[1] = *(const uint4*)(W2 + k0 + 8);
    b3.u[0] = *(const uint4*)(W3 + k0);
    b3.u[1] = *(const uint4*)(W3 + k0 + 8);
    if (k0 + 32 < K)
      __builtin_prefetch((const void*)(W0 + k0 + 32), 0, 1);
    acc[0] = __builtin_amdgcn_wmma_f32_16x16x32_bf16(
        false, a.v, false, b0.v, (short)0, acc[0], false, false);
    acc[1] = __builtin_amdgcn_wmma_f32_16x16x32_bf16(
        false, a.v, false, b1.v, (short)0, acc[1], false, false);
    acc[2] = __builtin_amdgcn_wmma_f32_16x16x32_bf16(
        false, a.v, false, b2.v, (short)0, acc[2], false, false);
    acc[3] = __builtin_amdgcn_wmma_f32_16x16x32_bf16(
        false, a.v, false, b3.v, (short)0, acc[3], false, false);
  }

  // C/D layout: lanes 0-15 N=lane, rows VGPRp -> M=p; lanes 16-31 -> M=p+8
  const int mout = mtile + ((lane >> 4) << 3);
  const int nlo  = lane & 15;
#pragma unroll
  for (int j = 0; j < 4; ++j) {
    const int col = ntile + j * 16 + nlo;
    const float bv = bias ? bias[col] : 0.0f;
#pragma unroll
    for (int p = 0; p < 8; ++p) {
      float v = acc[j][p] + bv;
      if (ACT == 1) v = (v > 0.0f) ? v : LEAK_ * v;     // leaky_relu
      const size_t off = (size_t)(mout + p) * ldc + col;
      if (Cf) Cf[off] = v;
      if (Cb) Cb[off] = f2bf(v);
    }
  }
}

// ---------------------------------------------------------------------------
// Elementwise / small kernels
// ---------------------------------------------------------------------------
__global__ void k_cvt_bf16(const float* __restrict__ s, u16* __restrict__ d, int n) {
  int i = blockIdx.x * blockDim.x + threadIdx.x;
  if (i < n) d[i] = f2bf(s[i]);
}
__global__ void k_fill_f32(float* p, float v, int n) {
  int i = blockIdx.x * blockDim.x + threadIdx.x;
  if (i < n) p[i] = v;
}
__global__ void k_fill_u16(u16* p, u16 v, int n) {
  int i = blockIdx.x * blockDim.x + threadIdx.x;
  if (i < n) p[i] = v;
}

// GRU gate: h2 = (1-z)*n + z*h ; gi/gh already include biases (added in GEMM)
__global__ void k_gru_gate(const float* __restrict__ gi, const float* __restrict__ gh,
                           const float* __restrict__ hin,
                           float* __restrict__ hf, u16* __restrict__ hb,
                           float* __restrict__ sf, u16* __restrict__ sb)
{
  int idx = blockIdx.x * blockDim.x + threadIdx.x;
  if (idx >= B_ * H_) return;
  int b = idx >> 9, j = idx & (H_ - 1);
  const float* gib = gi + (size_t)b * 3 * H_;
  const float* ghb = gh + (size_t)b * 3 * H_;
  float r = 1.0f / (1.0f + expf(-(gib[j]        + ghb[j])));
  float z = 1.0f / (1.0f + expf(-(gib[j + H_]   + ghb[j + H_])));
  float n = tanhf(gib[j + 2 * H_] + r * ghb[j + 2 * H_]);
  float h2 = (1.0f - z) * n + z * hin[idx];
  hf[idx] = h2;
  hb[idx] = f2bf(h2);
  if (sf) sf[idx] = h2;
  if (sb) sb[idx] = f2bf(h2);
}

__global__ void k_init_d(const float* __restrict__ embed, u16* __restrict__ db) {
  int i = blockIdx.x * blockDim.x + threadIdx.x;        // B*E
  if (i >= B_ * E_) return;
  db[i] = f2bf(embed[(size_t)BOS_ * E_ + (i & (E_ - 1))]);
}

// s[b,i] = tanh( dot(enc[i,b,:], h[b,:]) )
__global__ void k_attn_scores(const float* __restrict__ encf,
                              const float* __restrict__ h,
                              float* __restrict__ s)
{
  int idx = blockIdx.x * blockDim.x + threadIdx.x;
  if (idx >= B_ * NIN) return;
  int b = idx / NIN, i = idx % NIN;
  const float* e  = encf + (size_t)i * B_ * H_ + (size_t)b * H_;
  const float* hb = h + (size_t)b * H_;
  float acc = 0.0f;
  for (int k = 0; k < H_; ++k) acc += e[k] * hb[k];
  s[idx] = tanhf(acc);
}

// softmax over 80 scores, context c = sum_i w_i*enc[i,b,:], assemble x=[d|et|c] (bf16)
__global__ void k_attn_ctx(const float* __restrict__ encf,
                           const u16*  __restrict__ et_b,
                           const float* __restrict__ scores,
                           const u16*  __restrict__ d_b,
                           u16* __restrict__ x_b)
{
  int b = blockIdx.x, tid = threadIdx.x;                // 128 threads
  __shared__ float sw[NIN];
  __shared__ float red;
  if (tid == 0) {
    float m = -3.4e38f;
    for (int i = 0; i < NIN; ++i) m = fmaxf(m, scores[b * NIN + i]);
    red = m;
  }
  __syncthreads();
  float m = red;
  __syncthreads();
  if (tid < NIN) sw[tid] = expf(scores[b * NIN + tid] - m);
  __syncthreads();
  if (tid == 0) {
    float s = 0.0f;
    for (int i = 0; i < NIN; ++i) s += sw[i];
    red = s;
  }
  __syncthreads();
  float inv = 1.0f / red;
  for (int k = tid; k < H_; k += 128) {
    float c = 0.0f;
    for (int i = 0; i < NIN; ++i)
      c += sw[i] * encf[(size_t)i * B_ * H_ + (size_t)b * H_ + k];
    c *= inv;
    x_b[(size_t)b * 1536 + k]            = d_b[(size_t)b * E_ + k];   // d
    x_b[(size_t)b * 1536 + 512 + k]      = et_b[(size_t)b * H_ + k];  // enc_tail_t
    x_b[(size_t)b * 1536 + 1024 + k]     = f2bf(c);                   // context
  }
}

// per-row: argmax + log-softmax loss at tgt, and d_next = embed[argmax] (bf16)
__global__ __launch_bounds__(256) void k_loss_argmax(
    const float* __restrict__ logits, const int* __restrict__ tgt, int t,
    const float* __restrict__ embed, float* __restrict__ lossb, u16* __restrict__ db)
{
  int b = blockIdx.x, tid = threadIdx.x;
  const float* L = logits + (size_t)b * V_;
  __shared__ float sv[256];
  __shared__ int   si[256];
  float bm = -3.4e38f; int bi = V_;
  for (int j = tid; j < V_; j += 256) {
    float v = L[j];
    if (v > bm || (v == bm && j < bi)) { bm = v; bi = j; }
  }
  sv[tid] = bm; si[tid] = bi; __syncthreads();
  for (int s = 128; s > 0; s >>= 1) {
    if (tid < s) {
      if (sv[tid + s] > sv[tid] || (sv[tid + s] == sv[tid] && si[tid + s] < si[tid])) {
        sv[tid] = sv[tid + s]; si[tid] = si[tid + s];
      }
    }
    __syncthreads();
  }
  float gmax = sv[0];
  int gidx = si[0];
  __syncthreads();
  float se = 0.0f;
  for (int j = tid; j < V_; j += 256) se += expf(L[j] - gmax);
  sv[tid] = se; __syncthreads();
  for (int s = 128; s > 0; s >>= 1) { if (tid < s) sv[tid] += sv[tid + s]; __syncthreads(); }
  if (tid == 0) {
    int tj = tgt[b * NOUT + t];
    lossb[b] = -(L[tj] - gmax - logf(sv[0]));
  }
  for (int k = tid; k < E_; k += 256)
    db[(size_t)b * E_ + k] = f2bf(embed[(size_t)gidx * E_ + k]);
}

__global__ void k_loss_final(const float* __restrict__ lossm, float* __restrict__ out) {
  __shared__ float s[256];
  int tid = threadIdx.x;
  float a = 0.0f;
  for (int i = tid; i < NOUT * B_; i += 256) a += lossm[i];
  s[tid] = a; __syncthreads();
  for (int r = 128; r > 0; r >>= 1) { if (tid < r) s[tid] += s[tid + r]; __syncthreads(); }
  if (tid == 0) out[0] = s[0] / (float)(NOUT * B_);
}

// ---------------------------------------------------------------------------
static inline void launch_gemm(hipStream_t st, const u16* A, int lda,
                               const u16* W, int ldw, const float* bias,
                               float* Cf, u16* Cb, int ldc,
                               int M, int N, int K, int act)
{
  dim3 g((unsigned)(N / 64), (unsigned)((M + 127) / 128)), blk(256);
  if (act) k_gemm_wmma<1><<<g, blk, 0, st>>>(A, lda, W, ldw, bias, Cf, Cb, ldc, M, N, K);
  else     k_gemm_wmma<0><<<g, blk, 0, st>>>(A, lda, W, ldw, bias, Cf, Cb, ldc, M, N, K);
}

extern "C" void kernel_launch(void* const* d_in, const int* in_sizes, int n_in,
                              void* d_out, int out_size, void* d_ws, size_t ws_size,
                              hipStream_t stream)
{
  (void)in_sizes; (void)n_in; (void)out_size; (void)ws_size;
  const float* feats   = (const float*)d_in[0];
  const float* fc_w    = (const float*)d_in[1];
  const float* fc_b    = (const float*)d_in[2];
  const float* enc_wih = (const float*)d_in[3];
  const float* enc_whh = (const float*)d_in[4];
  const float* enc_bih = (const float*)d_in[5];
  const float* enc_bhh = (const float*)d_in[6];
  const float* dec_wih = (const float*)d_in[7];
  const float* dec_whh = (const float*)d_in[8];
  const float* dec_bih = (const float*)d_in[9];
  const float* dec_bhh = (const float*)d_in[10];
  const float* out_w   = (const float*)d_in[11];
  const float* out_b   = (const float*)d_in[12];
  const float* embed   = (const float*)d_in[13];
  const int*   tgt     = (const int*)d_in[14];

  // Workspace bump allocator (~305 MB total)
  char* w = (char*)d_ws;
  auto alloc = [&](size_t bytes) -> char* {
    char* p = w; w += (bytes + 255) & ~(size_t)255; return p;
  };
  u16*   feats_b = (u16*) alloc((size_t)NIN * B_ * F_ * 2);
  u16*   fcw_b  = (u16*)  alloc((size_t)E_ * F_ * 2);
  u16*   ewih_b = (u16*)  alloc((size_t)3 * H_ * E_ * 2);
  u16*   ewhh_b = (u16*)  alloc((size_t)3 * H_ * H_ * 2);
  u16*   dwih_b = (u16*)  alloc((size_t)3 * H_ * 1536 * 2);
  u16*   dwhh_b = (u16*)  alloc((size_t)3 * H_ * H_ * 2);
  u16*   outw_b = (u16*)  alloc((size_t)V_ * H_ * 2);
  u16*   xp_b   = (u16*)  alloc((size_t)TTOT * B_ * E_ * 2);
  float* gi_enc = (float*)alloc((size_t)TTOT * B_ * 3 * H_ * 4);
  float* enc_f  = (float*)alloc((size_t)TTOT * B_ * H_ * 4);
  u16*   enc_b  = (u16*)  alloc((size_t)TTOT * B_ * H_ * 2);
  float* gi_d1  = (float*)alloc((size_t)NIN * B_ * 3 * H_ * 4);
  float* h_f    = (float*)alloc((size_t)B_ * H_ * 4);
  u16*   h_b    = (u16*)  alloc((size_t)B_ * H_ * 2);
  float* gh_ws  = (float*)alloc((size_t)B_ * 3 * H_ * 4);
  float* gi_ws  = (float*)alloc((size_t)B_ * 3 * H_ * 4);
  u16*   x_b    = (u16*)  alloc((size_t)B_ * 1536 * 2);
  float* scores = (float*)alloc((size_t)B_ * NIN * 4);
  float* logits = (float*)alloc((size_t)B_ * V_ * 4);
  u16*   d_b    = (u16*)  alloc((size_t)B_ * E_ * 2);
  float* lossm  = (float*)alloc((size_t)NOUT * B_ * 4);

  auto cvt = [&](const float* s, u16* d, size_t n) {
    k_cvt_bf16<<<(unsigned)((n + 255) / 256), 256, 0, stream>>>(s, d, (int)n);
  };
  cvt(feats,   feats_b, (size_t)NIN * B_ * F_);
  cvt(fc_w,    fcw_b,  (size_t)E_ * F_);
  cvt(enc_wih, ewih_b, (size_t)3 * H_ * E_);
  cvt(enc_whh, ewhh_b, (size_t)3 * H_ * H_);
  cvt(dec_wih, dwih_b, (size_t)3 * H_ * 1536);
  cvt(dec_whh, dwhh_b, (size_t)3 * H_ * H_);
  cvt(out_w,   outw_b, (size_t)V_ * H_);

  // e_seq tail (steps NIN..TTOT) is zero
  {
    size_t n0 = (size_t)NIN * B_ * E_, n1 = (size_t)TTOT * B_ * E_;
    k_fill_u16<<<(unsigned)((n1 - n0 + 255) / 256), 256, 0, stream>>>(
        xp_b + n0, (u16)0, (int)(n1 - n0));
  }

  // FC: xp = leaky_relu(feats @ fc_w.T + fc_b)  -> bf16
  launch_gemm(stream, feats_b, F_, fcw_b, F_, fc_b,
              nullptr, xp_b, E_, NIN * B_, E_, F_, 1);

  // Encoder input-side gates for ALL 105 steps in one GEMM
  launch_gemm(stream, xp_b, E_, ewih_b, E_, enc_bih,
              gi_enc, nullptr, 3 * H_, TTOT * B_, 3 * H_, E_, 0);

  // Encoder recurrence
  k_fill_f32<<<(B_ * H_) / 256, 256, 0, stream>>>(h_f, 0.0f, B_ * H_);
  k_fill_u16<<<(B_ * H_) / 256, 256, 0, stream>>>(h_b, (u16)0, B_ * H_);
  for (int t = 0; t < TTOT; ++t) {
    launch_gemm(stream, h_b, H_, ewhh_b, H_, enc_bhh,
                gh_ws, nullptr, 3 * H_, B_, 3 * H_, H_, 0);
    k_gru_gate<<<(B_ * H_) / 256, 256, 0, stream>>>(
        gi_enc + (size_t)t * B_ * 3 * H_, gh_ws, h_f, h_f, h_b,
        enc_f + (size_t)t * B_ * H_, enc_b + (size_t)t * B_ * H_);
  }

  // dec1 input-side gates for all 80 steps: only last 512 cols of dec_wih matter
  launch_gemm(stream, enc_b, H_, dwih_b + (H_ + E_), 1536, dec_bih,
              gi_d1, nullptr, 3 * H_, NIN * B_, 3 * H_, H_, 0);

  // dec1 recurrence (h0 = 0)
  k_fill_f32<<<(B_ * H_) / 256, 256, 0, stream>>>(h_f, 0.0f, B_ * H_);
  k_fill_u16<<<(B_ * H_) / 256, 256, 0, stream>>>(h_b, (u16)0, B_ * H_);
  for (int t = 0; t < NIN; ++t) {
    launch_gemm(stream, h_b, H_, dwhh_b, H_, dec_bhh,
                gh_ws, nullptr, 3 * H_, B_, 3 * H_, H_, 0);
    k_gru_gate<<<(B_ * H_) / 256, 256, 0, stream>>>(
        gi_d1 + (size_t)t * B_ * 3 * H_, gh_ws, h_f, h_f, h_b, nullptr, nullptr);
  }

  // Attention decoder (h starts at alpha = dec1 final state, already in h_f/h_b)
  k_init_d<<<(B_ * E_) / 256, 256, 0, stream>>>(embed, d_b);
  for (int t = 0; t < NOUT; ++t) {
    k_attn_scores<<<(B_ * NIN + 255) / 256, 256, 0, stream>>>(enc_f, h_f, scores);
    k_attn_ctx<<<B_, 128, 0, stream>>>(
        enc_f, enc_b + (size_t)(NIN + t) * B_ * H_, scores, d_b, x_b);
    launch_gemm(stream, x_b, 1536, dwih_b, 1536, dec_bih,
                gi_ws, nullptr, 3 * H_, B_, 3 * H_, 1536, 0);
    launch_gemm(stream, h_b, H_, dwhh_b, H_, dec_bhh,
                gh_ws, nullptr, 3 * H_, B_, 3 * H_, H_, 0);
    k_gru_gate<<<(B_ * H_) / 256, 256, 0, stream>>>(
        gi_ws, gh_ws, h_f, h_f, h_b, nullptr, nullptr);
    launch_gemm(stream, h_b, H_, outw_b, H_, out_b,
                logits, nullptr, V_, B_, V_, H_, 0);
    k_loss_argmax<<<B_, 256, 0, stream>>>(logits, tgt, t, embed,
                                          lossm + (size_t)t * B_, d_b);
  }
  k_loss_final<<<1, 256, 0, stream>>>(lossm, (float*)d_out);
}
